// Codebook_47021301957004
// MI455X (gfx1250) — compile-verified, matrix operand order checked by reading, not run on hardware
//
#include <hip/hip_runtime.h>

// ---------------------------------------------------------------------------
// VQ codebook quantization for MI455X (gfx1250, wave32, WMMA).
//   z : [B=16, D=768, T=4096] fp32     e : [K=1024, D=768] fp32
// out: [ loss(1) | z_q [B,D,T] (50331648) | idx [B,T] (65536) ]  (floats)
// Math: argmin_k ||z - e_k||^2 == argmax_k (z . e_k - 0.5*||e_k||^2)
//
// Roofline: HBM floor ~405 MB => ~17 us @ 23.3 TB/s. GEMM = 103 GFLOP bf16
// WMMA. The binding secondary term is L2 codebook re-streaming; M=32 rows
// per block means each B fragment feeds 2 WMMAs (L2 traffic ~3 GB).
// B addressing uses uniform base + constant 32-bit lane offsets (GVS form),
// with ping-pong prefetch of the next B fragment across the 2 WMMAs.
// Workspace: bf16 codebook (1.5 MB) + 0.5*||e_k||^2 (4 KB).
// ---------------------------------------------------------------------------

#define B_DIM   16
#define D_DIM   768
#define T_DIM   4096
#define K_CODES 1024
#define M_ROWS  32                          // rows (t-values) per block
#define LDS_ROW (D_DIM + 8)                 // bf16 elems, pad to de-phase banks
#define ZQ_SIZE ((size_t)B_DIM * D_DIM * T_DIM)
#define IDX_OFF (1 + ZQ_SIZE)

typedef __attribute__((ext_vector_type(16))) __bf16 v16bf;
typedef __attribute__((ext_vector_type(8)))  __bf16 v8bf;
typedef __attribute__((ext_vector_type(8)))  float  v8f;

union BF16Frag { v16bf v; v8bf h[2]; };

// ---------------------------------------------------------------------------
// Prep: e fp32 -> bf16 codebook in ws, half-norms, zero the loss accumulator.
// ---------------------------------------------------------------------------
__global__ __launch_bounds__(256) void vq_prep(const float* __restrict__ e,
                                               __bf16* __restrict__ eb,
                                               float* __restrict__ hn,
                                               float* __restrict__ out) {
  const int k   = blockIdx.x;
  const int tid = threadIdx.x;
  if (k == 0 && tid == 0) out[0] = 0.0f;    // loss accumulator (main adds)

  const float* er  = e  + (size_t)k * D_DIM;
  __bf16*      ebr = eb + (size_t)k * D_DIM;
  float s = 0.0f;
  for (int d = tid; d < D_DIM; d += 256) {
    float v = er[d];
    s += v * v;
    ebr[d] = (__bf16)v;
  }
  #pragma unroll
  for (int off = 16; off >= 1; off >>= 1) s += __shfl_xor(s, off, 32);
  __shared__ float red[8];
  if ((tid & 31) == 0) red[tid >> 5] = s;
  __syncthreads();
  if (tid == 0) {
    float t = 0.0f;
    #pragma unroll
    for (int w = 0; w < 8; ++w) t += red[w];
    hn[k] = 0.5f * t;
  }
}

// ---------------------------------------------------------------------------
// Main: one block = 32 rows of zf (32 consecutive t of one b) vs all K codes.
// 256 threads = 8 waves; wave w owns codes [w*128, w*128+128) as 8 N-tiles,
// and both 16-row M-tiles (so each B fragment is reused by 2 WMMAs).
// ---------------------------------------------------------------------------
__global__ __launch_bounds__(256, 1) void vq_main(const float* __restrict__ z,
                                                  const float* __restrict__ e,
                                                  const __bf16* __restrict__ eb,
                                                  const float* __restrict__ hn,
                                                  float* __restrict__ out) {
  __shared__ __align__(16) __bf16 zb[M_ROWS * LDS_ROW];  // A tiles, bf16
  __shared__ float red_s[8 * M_ROWS];
  __shared__ int   red_i[8 * M_ROWS];
  __shared__ int   fin_idx[M_ROWS];
  __shared__ float lred[8];

  const int tid  = threadIdx.x;
  const int wave = tid >> 5;
  const int lane = tid & 31;
  const int hgrp = lane >> 4;                 // 0: K-lo half, 1: K-hi half
  const int l16  = lane & 15;

  const int rbase = blockIdx.x * M_ROWS;      // row in flattened [B*T]
  const int bidx  = rbase / T_DIM;
  const int t0    = rbase % T_DIM;
  const float* zB = z + (size_t)bidx * D_DIM * T_DIM + t0;

  // ---- stage z tile into LDS as bf16: zb[m][d], m = t offset 0..31 --------
  for (int it = tid; it < D_DIM * (M_ROWS / 4); it += 256) {
    const int d    = it >> 3;
    const int part = it & 7;
    const float4 v = *(const float4*)(zB + (size_t)d * T_DIM + part * 4);
    const int m = part * 4;
    zb[(m + 0) * LDS_ROW + d] = (__bf16)v.x;
    zb[(m + 1) * LDS_ROW + d] = (__bf16)v.y;
    zb[(m + 2) * LDS_ROW + d] = (__bf16)v.z;
    zb[(m + 3) * LDS_ROW + d] = (__bf16)v.w;
  }
  __syncthreads();

  // ---- WMMA: scores[32 rows x 128 codes] per wave, K-dim 768 = 24 x 32 ----
  const int nbase = wave * 128;
  v8f acc0[8] = {};                           // M-tile 0 (rows 0..15)
  v8f acc1[8] = {};                           // M-tile 1 (rows 16..31)

  // Per-lane constant 32-bit B offsets (elements): code row + K-half select.
  int boff[8];
  #pragma unroll
  for (int nt = 0; nt < 8; ++nt)
    boff[nt] = (nbase + nt * 16 + l16) * D_DIM + hgrp * 8;

  // A-fragment per-lane sources: row l16 (+16), K-halves split by lane group
  const __bf16* aBase0 = &zb[l16 * LDS_ROW + hgrp * 8];
  const __bf16* aBase1 = aBase0 + 16 * LDS_ROW;

  for (int kc = 0; kc < D_DIM; kc += 32) {
    BF16Frag A0, A1;
    A0.h[0] = *(const v8bf*)(aBase0 + kc);      // K = kc + hgrp*8 .. +7
    A0.h[1] = *(const v8bf*)(aBase0 + kc + 16); // K = kc+16+hgrp*8 .. +7
    A1.h[0] = *(const v8bf*)(aBase1 + kc);
    A1.h[1] = *(const v8bf*)(aBase1 + kc + 16);

    const __bf16* bk = eb + kc;                 // uniform base (saddr form)
    BF16Frag Bc, Bn;
    Bc.h[0] = *(const v8bf*)(bk + boff[0]);
    Bc.h[1] = *(const v8bf*)(bk + boff[0] + 16);
    #pragma unroll
    for (int nt = 0; nt < 8; ++nt) {
      if (nt < 7) {                             // prefetch next B fragment
        Bn.h[0] = *(const v8bf*)(bk + boff[nt + 1]);
        Bn.h[1] = *(const v8bf*)(bk + boff[nt + 1] + 16);
      }
      acc0[nt] = __builtin_amdgcn_wmma_f32_16x16x32_bf16(
          false, A0.v, false, Bc.v, (short)0, acc0[nt], false, false);
      acc1[nt] = __builtin_amdgcn_wmma_f32_16x16x32_bf16(
          false, A1.v, false, Bc.v, (short)0, acc1[nt], false, false);
      Bc = Bn;
    }
  }

  // ---- per-lane argmax of (z.e - 0.5||e||^2), tie -> lowest index ---------
  float best[2][8];
  int   bidxv[2][8];
  #pragma unroll
  for (int mt = 0; mt < 2; ++mt)
    #pragma unroll
    for (int vi = 0; vi < 8; ++vi) {
      best[mt][vi] = -3.0e38f; bidxv[mt][vi] = 0x7fffffff;
    }
  #pragma unroll
  for (int nt = 0; nt < 8; ++nt) {
    const int   n = nbase + nt * 16 + l16;
    const float h = hn[n];
    #pragma unroll
    for (int vi = 0; vi < 8; ++vi) {
      const float s0 = acc0[nt][vi] - h;
      if (s0 > best[0][vi] || (s0 == best[0][vi] && n < bidxv[0][vi])) {
        best[0][vi] = s0; bidxv[0][vi] = n;
      }
      const float s1 = acc1[nt][vi] - h;
      if (s1 > best[1][vi] || (s1 == best[1][vi] && n < bidxv[1][vi])) {
        best[1][vi] = s1; bidxv[1][vi] = n;
      }
    }
  }
  // C-layout: lanes 0-15 hold rows vi, lanes 16-31 rows vi+8 -> reduce
  // across the 16 lanes of each half (xor masks stay inside the half).
  #pragma unroll
  for (int off = 1; off < 16; off <<= 1) {
    #pragma unroll
    for (int mt = 0; mt < 2; ++mt)
      #pragma unroll
      for (int vi = 0; vi < 8; ++vi) {
        const float os = __shfl_xor(best[mt][vi], off, 32);
        const int   oi = __shfl_xor(bidxv[mt][vi], off, 32);
        if (os > best[mt][vi] || (os == best[mt][vi] && oi < bidxv[mt][vi])) {
          best[mt][vi] = os; bidxv[mt][vi] = oi;
        }
      }
  }
  if (l16 == 0) {
    #pragma unroll
    for (int mt = 0; mt < 2; ++mt)
      #pragma unroll
      for (int vi = 0; vi < 8; ++vi) {
        const int row = mt * 16 + vi + hgrp * 8;
        red_s[wave * M_ROWS + row] = best[mt][vi];
        red_i[wave * M_ROWS + row] = bidxv[mt][vi];
      }
  }
  __syncthreads();

  // ---- cross-wave reduce, write indices -----------------------------------
  if (tid < M_ROWS) {
    float bs = red_s[tid];
    int   bi = red_i[tid];
    #pragma unroll
    for (int w = 1; w < 8; ++w) {
      const float s  = red_s[w * M_ROWS + tid];
      const int   i2 = red_i[w * M_ROWS + tid];
      if (s > bs || (s == bs && i2 < bi)) { bs = s; bi = i2; }
    }
    fin_idx[tid] = bi;
    out[IDX_OFF + rbase + tid] = (float)bi;
  }
  __syncthreads();

  // ---- gather fp32 code rows, write z_q [B,D,T], accumulate loss ----------
  float lsum = 0.0f;
  float* zq = out + 1 + (size_t)bidx * D_DIM * T_DIM + t0;
  for (int it = tid; it < D_DIM * M_ROWS; it += 256) {
    const int d = it >> 5;
    const int m = it & 31;                 // 32 consecutive t -> 128B stores
    const float ev = e[(size_t)fin_idx[m] * D_DIM + d];
    const float zv = zB[(size_t)d * T_DIM + m];
    const float df = zv - ev;
    lsum += df * df;
    zq[(size_t)d * T_DIM + m] = ev;
  }
  #pragma unroll
  for (int off = 16; off >= 1; off >>= 1) lsum += __shfl_xor(lsum, off, 32);
  if (lane == 0) lred[wave] = lsum;
  __syncthreads();
  if (tid == 0) {
    float s = 0.0f;
    #pragma unroll
    for (int w = 0; w < 8; ++w) s += lred[w];
    atomicAdd(out, s * (1.0f / (float)((size_t)B_DIM * T_DIM * D_DIM)));
  }
}

// ---------------------------------------------------------------------------
extern "C" void kernel_launch(void* const* d_in, const int* in_sizes, int n_in,
                              void* d_out, int out_size, void* d_ws, size_t ws_size,
                              hipStream_t stream) {
  (void)in_sizes; (void)n_in; (void)out_size; (void)ws_size;
  const float* z = (const float*)d_in[0];
  const float* e = (const float*)d_in[1];
  float* out = (float*)d_out;
  // ws: [ bf16 codebook 1.5MB | half-norms 4KB ]
  __bf16* eb = (__bf16*)d_ws;
  float*  hn = (float*)((char*)d_ws + (size_t)K_CODES * D_DIM * sizeof(__bf16));

  vq_prep<<<K_CODES, 256, 0, stream>>>(e, eb, hn, out);
  vq_main<<<(B_DIM * T_DIM) / M_ROWS, 256, 0, stream>>>(z, e, eb, hn, out);
}
